// FrequencyGatedSSM_77549929497027
// MI455X (gfx1250) — matrix operand order repaired and checked by reading, not compiled, not run on hardware
//
#include <hip/hip_runtime.h>

// ---------------------------------------------------------------------------
// FrequencyGatedSSM fused kernel for gfx1250 (MI455X), wave32 + WMMA f16.
// B=8, L=65536, C=D=64. One block = 128 rows of one batch; 8 waves.
// Dual-subtile interleaved WMMA to fill WMMA->WMMA hazard slots.
// ---------------------------------------------------------------------------

typedef __attribute__((ext_vector_type(16))) _Float16 v16h;
typedef __attribute__((ext_vector_type(8)))  _Float16 v8h;
typedef __attribute__((ext_vector_type(8)))  float    v8f;

#define LSEQ      65536
#define NBATCH    8
#define TILE      128
#define CHUNKS    (LSEQ / TILE)        // 512
#define NTHREADS  256
#define SW        72                   // padded f16 row stride (halves)

// LDS layout (halves then floats)
#define H_WINT    0
#define H_WIPT    (H_WINT  + 64  * SW)
#define H_WOPT    (H_WIPT  + 128 * SW)
#define H_WOUTT   (H_WOPT  + 64  * SW)
#define H_BUFA    (H_WOUTT + 64  * SW)   // 128*SW : xA -> RES
#define H_BUFB    (H_BUFA  + 128 * SW)   // 128*SW : featA -> HA
#define H_BUFC    (H_BUFB  + 128 * SW)   // 130*SW : XB (halo rows 0,129) -> F2
#define H_TOTAL   (H_BUFC  + 130 * SW)   // 50832 halves

#define F_G16     0                       // 128*16
#define F_WG2     (F_G16   + 128*16)      // 16*64
#define F_BIN     (F_WG2   + 16*64)       // 64
#define F_BIP     (F_BIN   + 64)          // 128
#define F_CW      (F_BIP   + 128)         // 192
#define F_CB      (F_CW    + 192)         // 64
#define F_BOP     (F_CB    + 64)          // 64
#define F_BOUT    (F_BOP   + 64)          // 64
#define F_BG2     (F_BOUT  + 64)          // 64
#define F_WG1     (F_BG2   + 64)          // 16
#define F_BG1     (F_WG1   + 16)          // 16
#define F_FREQ    (F_BG1   + 16)          // 128
#define F_FEATH   (F_FREQ  + 128)         // 2*64
#define F_XH      (F_FEATH + 128)         // 2*64
#define F_TOTAL   (F_XH    + 128)         // 4128 floats

#define SMEM_BYTES (H_TOTAL * 2 + F_TOTAL * 4)   // 118176

#define WMMA_F16(A, B, C) \
  __builtin_amdgcn_wmma_f32_16x16x32_f16(false, (A), false, (B), (short)0, (C), false, false)

__device__ __forceinline__ float silu_f(float v) {
  return v / (1.0f + __expf(-v));
}

// A fragment: 16x32 f16 from LDS tile (row-major, stride SW halves).
__device__ __forceinline__ v16h load_frag_a(const _Float16* base, int rowBase,
                                            int k0, int lane) {
  int r = lane & 15, cs = lane >> 4;
  const _Float16* p = base + (rowBase + r) * SW + k0 + 8 * cs;
  v8h lo = *(const v8h*)p;
  v8h hi = *(const v8h*)(p + 16);
  return __builtin_shufflevector(lo, hi, 0, 1, 2, 3, 4, 5, 6, 7,
                                 8, 9, 10, 11, 12, 13, 14, 15);
}

// B fragment: 32x16 f16 from transposed weights WT[n][k] (stride SW halves).
__device__ __forceinline__ v16h load_frag_b(const _Float16* wt, int n0,
                                            int k0, int lane) {
  int r = lane & 15, cs = lane >> 4;
  const _Float16* p = wt + (n0 + r) * SW + k0 + 8 * cs;
  v8h lo = *(const v8h*)p;
  v8h hi = *(const v8h*)(p + 16);
  return __builtin_shufflevector(lo, hi, 0, 1, 2, 3, 4, 5, 6, 7,
                                 8, 9, 10, 11, 12, 13, 14, 15);
}

// D fragment (f32, 8 VGPRs) -> f16 LDS tile. row = rowBase + 8*cs + r, col = n0 + (lane&15)
__device__ __forceinline__ void store_frag_f16(_Float16* base, int rowBase,
                                               int n0, int lane, v8f acc) {
  int c = lane & 15, cs = lane >> 4;
#pragma unroll
  for (int r = 0; r < 8; ++r)
    base[(rowBase + 8 * cs + r) * SW + n0 + c] = (_Float16)acc[r];
}

__device__ __forceinline__ v8f bias_frag(const float* bias, int n0, int lane) {
  float bv = bias[n0 + (lane & 15)];
  v8f acc = {bv, bv, bv, bv, bv, bv, bv, bv};
  return acc;
}

// Two independent 16-col subtiles computed with interleaved WMMAs (fills the
// WMMA->WMMA hazard slots with independent work), results stored as f16.
//   nb0 : B-matrix / bias column base;  dn0 : destination column base.
__device__ __forceinline__ void gemm_pair_store(const _Float16* WT, const float* bias,
                                                int nb0, v16h a0, v16h a1,
                                                _Float16* dst, int dstRowBase,
                                                int dn0, int lane) {
  v8f acc0 = bias_frag(bias, nb0, lane);
  v8f acc1 = bias_frag(bias, nb0 + 16, lane);
  v16h b00 = load_frag_b(WT, nb0,      0,  lane);
  v16h b01 = load_frag_b(WT, nb0 + 16, 0,  lane);
  v16h b10 = load_frag_b(WT, nb0,      32, lane);
  v16h b11 = load_frag_b(WT, nb0 + 16, 32, lane);
  acc0 = WMMA_F16(a0, b00, acc0);
  acc1 = WMMA_F16(a0, b01, acc1);
  acc0 = WMMA_F16(a1, b10, acc0);
  acc1 = WMMA_F16(a1, b11, acc1);
  store_frag_f16(dst, dstRowBase, dn0,      lane, acc0);
  store_frag_f16(dst, dstRowBase, dn0 + 16, lane, acc1);
}

__global__ __launch_bounds__(NTHREADS, 2)
void fgssm_fused(const float* __restrict__ x, const float* __restrict__ freq,
                 const float* __restrict__ W_in, const float* __restrict__ b_in,
                 const float* __restrict__ W_ip, const float* __restrict__ b_ip,
                 const float* __restrict__ conv_w, const float* __restrict__ conv_b,
                 const float* __restrict__ W_op, const float* __restrict__ b_op,
                 const float* __restrict__ W_g1, const float* __restrict__ b_g1,
                 const float* __restrict__ W_g2, const float* __restrict__ b_g2,
                 const float* __restrict__ W_out, const float* __restrict__ b_out,
                 float* __restrict__ out) {
  extern __shared__ __align__(16) char smem[];
  _Float16* hbase = (_Float16*)smem;
  float*    fbase = (float*)(smem + H_TOTAL * 2);

  _Float16* WinT  = hbase + H_WINT;
  _Float16* WipT  = hbase + H_WIPT;
  _Float16* WopT  = hbase + H_WOPT;
  _Float16* WoutT = hbase + H_WOUTT;
  _Float16* bufA  = hbase + H_BUFA;   // xA, later RES
  _Float16* bufB  = hbase + H_BUFB;   // featA, later HA
  _Float16* bufC  = hbase + H_BUFC;   // XB (rows 0..129), later F2 (rows 0..127)

  float* g16    = fbase + F_G16;
  float* Wg2s   = fbase + F_WG2;
  float* b_in_s = fbase + F_BIN;
  float* b_ip_s = fbase + F_BIP;
  float* cw_s   = fbase + F_CW;
  float* cb_s   = fbase + F_CB;
  float* b_op_s = fbase + F_BOP;
  float* b_out_s= fbase + F_BOUT;
  float* b_g2_s = fbase + F_BG2;
  float* Wg1_s  = fbase + F_WG1;
  float* bg1_s  = fbase + F_BG1;
  float* fqs    = fbase + F_FREQ;
  float* feat_h = fbase + F_FEATH;
  float* xh     = fbase + F_XH;

  const int tid   = threadIdx.x;
  const int lane  = tid & 31;
  const int wave  = tid >> 5;
  const int rowBase = wave * 16;
  const int c  = lane & 15;
  const int cs = lane >> 4;

  const int bb    = blockIdx.x / CHUNKS;
  const int chunk = blockIdx.x % CHUNKS;
  const size_t rowG = (size_t)bb * LSEQ + (size_t)chunk * TILE;

  // ---------------- Phase 1: stage weights (f16, transposed) + x tile -------
  for (int i = tid; i < 64 * 64; i += NTHREADS) {
    int k = i >> 6, n = i & 63;
    WinT [n * SW + k] = (_Float16)W_in [i];
    WopT [n * SW + k] = (_Float16)W_op [i];
    WoutT[n * SW + k] = (_Float16)W_out[i];
  }
  for (int i = tid; i < 64 * 128; i += NTHREADS) {
    int k = i >> 7, n = i & 127;
    WipT[n * SW + k] = (_Float16)W_ip[i];
  }
  for (int i = tid; i < 16 * 64; i += NTHREADS) Wg2s[i] = W_g2[i];
  if (tid < 64) {
    b_in_s[tid] = b_in[tid];   cb_s[tid]   = conv_b[tid];
    b_op_s[tid] = b_op[tid];   b_out_s[tid]= b_out[tid];
    b_g2_s[tid] = b_g2[tid];
  }
  if (tid < 128) b_ip_s[tid] = b_ip[tid];
  if (tid >= 128 && tid < 128 + 192) cw_s[tid - 128] = conv_w[tid - 128];
  if (tid < 16) { Wg1_s[tid] = W_g1[tid]; bg1_s[tid] = b_g1[tid]; }
  for (int i = tid; i < TILE; i += NTHREADS) fqs[i] = freq[rowG + i];
  for (int i = tid; i < TILE * 64; i += NTHREADS) {
    int l = i >> 6, k = i & 63;
    bufA[l * SW + k] = (_Float16)x[(rowG + l) * 64 + k];
  }
  if (tid < 128) {   // halo x rows (-1 and TILE), zero at batch edges
    int hs = tid >> 6, j = tid & 63;
    long hl = hs == 0 ? (long)chunk * TILE - 1 : (long)chunk * TILE + TILE;
    float v = 0.0f;
    if (hl >= 0 && hl < LSEQ) v = x[((size_t)bb * LSEQ + hl) * 64 + j];
    xh[hs * 64 + j] = v;
  }
  __syncthreads();

  // ---------------- Phase 2: feat = x @ W_in + b_in  (-> bufB f16) ----------
  {
    v16h a0 = load_frag_a(bufA, rowBase, 0,  lane);
    v16h a1 = load_frag_a(bufA, rowBase, 32, lane);
    gemm_pair_store(WinT, b_in_s, 0,  a0, a1, bufB, rowBase, 0,  lane);
    gemm_pair_store(WinT, b_in_s, 32, a0, a1, bufB, rowBase, 32, lane);
  }
  if (tid < 128) {   // halo feat rows (scalar, from staged xh + f16 weights)
    int hs = tid >> 6, j = tid & 63;
    float a = b_in_s[j];
#pragma unroll 8
    for (int k = 0; k < 64; ++k) a += xh[hs * 64 + k] * (float)WinT[j * SW + k];
    feat_h[hs * 64 + j] = a;
  }
  __syncthreads();

  // ---------------- Phase 3: x_and_res = feat @ W_ip + b_ip -----------------
  //   n<64  -> x_branch  -> bufC rows 1..128
  //   n>=64 -> res       -> bufA rows 0..127
  {
    v16h a0 = load_frag_a(bufB, rowBase, 0,  lane);
    v16h a1 = load_frag_a(bufB, rowBase, 32, lane);
    gemm_pair_store(WipT, b_ip_s, 0,  a0, a1, bufC, rowBase + 1, 0,  lane);
    gemm_pair_store(WipT, b_ip_s, 32, a0, a1, bufC, rowBase + 1, 32, lane);
    gemm_pair_store(WipT, b_ip_s, 64, a0, a1, bufA, rowBase,     0,  lane);
    gemm_pair_store(WipT, b_ip_s, 96, a0, a1, bufA, rowBase,     32, lane);
  }
  if (tid < 128) {   // halo x_branch rows 0 / 129 (zero at batch edges)
    int hs = tid >> 6, j = tid & 63;
    bool oob = (hs == 0) ? (chunk == 0) : (chunk == CHUNKS - 1);
    float v = 0.0f;
    if (!oob) {
      v = b_ip_s[j];
#pragma unroll 8
      for (int k = 0; k < 64; ++k) v += feat_h[hs * 64 + k] * (float)WipT[j * SW + k];
    }
    bufC[(hs ? (TILE + 1) : 0) * SW + j] = (_Float16)v;
  }
  __syncthreads();

  // ---------------- Phase 4: depthwise conv3 + SiLU*SiLU + gate MLP pt1 -----
  for (int i = tid; i < TILE * 64; i += NTHREADS) {
    int l = i >> 6, d = i & 63;
    float xm = (float)bufC[(l    ) * SW + d];
    float x0 = (float)bufC[(l + 1) * SW + d];
    float xp = (float)bufC[(l + 2) * SW + d];
    float cv = cw_s[d * 3] * xm + cw_s[d * 3 + 1] * x0 + cw_s[d * 3 + 2] * xp + cb_s[d];
    float rv = (float)bufA[l * SW + d];
    bufB[l * SW + d] = (_Float16)(silu_f(cv) * silu_f(rv));
  }
  for (int i = tid; i < TILE * 16; i += NTHREADS) {
    int l = i >> 4, j = i & 15;
    float t = fqs[l] * Wg1_s[j] + bg1_s[j];
    g16[i] = t > 0.0f ? t : 0.0f;
  }
  __syncthreads();

  // ---------------- Phase 5: feat2 = h @ W_op + b_op  (-> bufC f16) ---------
  {
    v16h a0 = load_frag_a(bufB, rowBase, 0,  lane);
    v16h a1 = load_frag_a(bufB, rowBase, 32, lane);
    gemm_pair_store(WopT, b_op_s, 0,  a0, a1, bufC, rowBase, 0,  lane);
    gemm_pair_store(WopT, b_op_s, 32, a0, a1, bufC, rowBase, 32, lane);
  }
  __syncthreads();

  // ---------------- Phase 6: out = (feat2 @ W_out + b_out) * gate -----------
  {
    v16h a0 = load_frag_a(bufC, rowBase, 0,  lane);
    v16h a1 = load_frag_a(bufC, rowBase, 32, lane);
#pragma unroll
    for (int sp = 0; sp < 4; sp += 2) {
      v8f acc0 = bias_frag(b_out_s, sp * 16,      lane);
      v8f acc1 = bias_frag(b_out_s, sp * 16 + 16, lane);
      v16h b00 = load_frag_b(WoutT, sp * 16,      0,  lane);
      v16h b01 = load_frag_b(WoutT, sp * 16 + 16, 0,  lane);
      v16h b10 = load_frag_b(WoutT, sp * 16,      32, lane);
      v16h b11 = load_frag_b(WoutT, sp * 16 + 16, 32, lane);
      acc0 = WMMA_F16(a0, b00, acc0);
      acc1 = WMMA_F16(a0, b01, acc1);
      acc0 = WMMA_F16(a1, b10, acc0);
      acc1 = WMMA_F16(a1, b11, acc1);
#pragma unroll
      for (int half = 0; half < 2; ++half) {
        v8f acc = half ? acc1 : acc0;
        int col = (sp + half) * 16 + c;
        float bg = b_g2_s[col];
#pragma unroll
        for (int r = 0; r < 8; ++r) {
          int lrow = rowBase + 8 * cs + r;
          float z = bg;
#pragma unroll
          for (int j = 0; j < 16; ++j) z += g16[lrow * 16 + j] * Wg2s[j * 64 + col];
          float gate = 1.0f / (1.0f + __expf(-z));
          out[(rowG + lrow) * 64 + col] = acc[r] * gate;
        }
      }
    }
  }
}

extern "C" void kernel_launch(void* const* d_in, const int* in_sizes, int n_in,
                              void* d_out, int out_size, void* d_ws, size_t ws_size,
                              hipStream_t stream) {
  (void)in_sizes; (void)n_in; (void)out_size; (void)d_ws; (void)ws_size;
  const float* x      = (const float*)d_in[0];
  const float* freq   = (const float*)d_in[1];
  const float* W_in   = (const float*)d_in[2];
  const float* b_in   = (const float*)d_in[3];
  const float* W_ip   = (const float*)d_in[4];
  const float* b_ip   = (const float*)d_in[5];
  const float* conv_w = (const float*)d_in[6];
  const float* conv_b = (const float*)d_in[7];
  const float* W_op   = (const float*)d_in[8];
  const float* b_op   = (const float*)d_in[9];
  const float* W_g1   = (const float*)d_in[10];
  const float* b_g1   = (const float*)d_in[11];
  const float* W_g2   = (const float*)d_in[12];
  const float* b_g2   = (const float*)d_in[13];
  const float* W_out  = (const float*)d_in[14];
  const float* b_out  = (const float*)d_in[15];
  float* out = (float*)d_out;

  (void)hipFuncSetAttribute((const void*)fgssm_fused,
                            hipFuncAttributeMaxDynamicSharedMemorySize,
                            SMEM_BYTES);

  dim3 grid(NBATCH * CHUNKS);   // 4096 blocks
  dim3 block(NTHREADS);         // 8 wave32s
  fgssm_fused<<<grid, block, SMEM_BYTES, stream>>>(
      x, freq, W_in, b_in, W_ip, b_ip, conv_w, conv_b,
      W_op, b_op, W_g1, b_g1, W_g2, b_g2, W_out, b_out, out);
}
